// PartialFC_66907000537296
// MI455X (gfx1250) — matrix operand order repaired and checked by reading, not compiled, not run on hardware
//
#include <hip/hip_runtime.h>
#include <hip/hip_bf16.h>
#include <math.h>

typedef _Float16 v16h __attribute__((ext_vector_type(16)));
typedef _Float16 v8h  __attribute__((ext_vector_type(8)));
typedef float    v8f  __attribute__((ext_vector_type(8)));
typedef unsigned int u32x4 __attribute__((ext_vector_type(4)));
typedef int          i32x8 __attribute__((ext_vector_type(8)));
typedef int          i32x4 __attribute__((ext_vector_type(4)));

#define NC     1000000   // num classes
#define NB     3907      // ceil(NC/256)
#define BATCH  4096
#define DIM    128
#define WTOT   100000    // n + num_sample == 100000 always (n <= 4096)
#define WPAD   100032    // WTOT padded to a multiple of 64 (branch-free GEMM)

// ---------------- unique(y) pipeline ----------------

__global__ void k_clear(unsigned int* flags) {
    int g = blockIdx.x * 256 + threadIdx.x;
    if (g < NC) flags[g] = 0u;
}

__global__ void k_mark(const int* y, unsigned int* flags) {
    int g = blockIdx.x * 256 + threadIdx.x;
    if (g < BATCH) flags[y[g]] = 1u;
}

__global__ void k_block_reduce(const unsigned int* flags, unsigned int* bsum) {
    __shared__ unsigned int s[256];
    int t = threadIdx.x;
    int g = blockIdx.x * 256 + t;
    s[t] = (g < NC) ? flags[g] : 0u;
    __syncthreads();
    for (int off = 128; off > 0; off >>= 1) {
        if (t < off) s[t] += s[t + off];
        __syncthreads();
    }
    if (t == 0) bsum[blockIdx.x] = s[0];
}

__global__ void k_scan_bsums(unsigned int* bsum, int* d_n) {
    __shared__ unsigned int s[256];
    __shared__ unsigned int carry;
    int t = threadIdx.x;
    if (t == 0) carry = 0u;
    __syncthreads();
    for (int base = 0; base < NB; base += 256) {
        unsigned int v = (base + t < NB) ? bsum[base + t] : 0u;
        s[t] = v; __syncthreads();
        for (int off = 1; off < 256; off <<= 1) {
            unsigned int u = 0; if (t >= off) u = s[t - off];
            __syncthreads(); s[t] += u; __syncthreads();
        }
        unsigned int ex = s[t] - v + carry;      // exclusive + running carry
        if (base + t < NB) bsum[base + t] = ex;
        __syncthreads();
        if (t == 0) carry += s[255];
        __syncthreads();
    }
    if (t == 0) *d_n = (int)carry;               // n = #unique labels
}

__global__ void k_emit_pos(const unsigned int* flags, const unsigned int* bsum,
                           int* positive, int* idx_map) {
    __shared__ unsigned int s[256];
    int t = threadIdx.x;
    int g = blockIdx.x * 256 + t;
    unsigned int f = (g < NC) ? flags[g] : 0u;
    s[t] = f; __syncthreads();
    for (int off = 1; off < 256; off <<= 1) {
        unsigned int u = 0; if (t >= off) u = s[t - off];
        __syncthreads(); s[t] += u; __syncthreads();
    }
    unsigned int ex = s[t] - f;
    if (g < NC && f) {
        unsigned int r = bsum[blockIdx.x] + ex;
        positive[r] = g;
        idx_map[g] = (int)r;
    }
}

__global__ void k_remap(const int* y, const int* idx_map, float* out_tail) {
    int g = blockIdx.x * 256 + threadIdx.x;
    if (g < BATCH) out_tail[g] = (float)idx_map[y[g]];
}

// ---------------- descending stable radix sort of perm (top-k) ----------------

__global__ void k_init_keys(const float* perm, const unsigned int* flags,
                            unsigned int* key, unsigned int* pay) {
    int g = blockIdx.x * 256 + threadIdx.x;
    if (g < NC) {
        float f = flags[g] ? -1.0f : perm[g];          // mask positives
        unsigned int b = __float_as_uint(f);
        unsigned int asc = b ^ ((b & 0x80000000u) ? 0xFFFFFFFFu : 0x80000000u);
        key[g] = ~asc;                                  // ascending key == descending f
        pay[g] = (unsigned int)g;
    }
}

__global__ void k_hist(const unsigned int* key, unsigned int* hist, int shift) {
    __shared__ unsigned int h[256];
    int t = threadIdx.x;
    h[t] = 0u; __syncthreads();
    int g = blockIdx.x * 256 + t;
    if (g < NC) atomicAdd(&h[(key[g] >> shift) & 255u], 1u);
    __syncthreads();
    hist[(size_t)blockIdx.x * 256 + t] = h[t];
}

__global__ void k_scan_hist(unsigned int* hist) {   // 1 block, 256 threads
    int d = threadIdx.x;
    unsigned int run = 0u;
    for (int b = 0; b < NB; ++b) {                   // per-digit scan over blocks
        size_t i = (size_t)b * 256 + d;
        unsigned int c = hist[i]; hist[i] = run; run += c;
    }
    __shared__ unsigned int s[256];
    s[d] = run; __syncthreads();
    for (int off = 1; off < 256; off <<= 1) {
        unsigned int u = 0; if (d >= off) u = s[d - off];
        __syncthreads(); s[d] += u; __syncthreads();
    }
    unsigned int base = s[d] - run;                  // exclusive digit base
    for (int b = 0; b < NB; ++b) hist[(size_t)b * 256 + d] += base;
}

__global__ void k_scatter(const unsigned int* keyIn, const unsigned int* payIn,
                          unsigned int* keyOut, unsigned int* payOut,
                          const unsigned int* hist, int shift) {
    __shared__ unsigned int skey[256], spay[256], sscan[256];
    int t = threadIdx.x;
    int g = blockIdx.x * 256 + t;
    unsigned int k = (g < NC) ? keyIn[g] : 0xFFFFFFFFu;  // sentinel sorts last
    unsigned int p = (g < NC) ? payIn[g] : 0u;
    // 8 rounds of stable 1-bit split -> block locally sorted by this digit
    for (int b = 0; b < 8; ++b) {
        unsigned int bit = (k >> (shift + b)) & 1u;
        sscan[t] = bit; __syncthreads();
        for (int off = 1; off < 256; off <<= 1) {
            unsigned int u = 0; if (t >= off) u = sscan[t - off];
            __syncthreads(); sscan[t] += u; __syncthreads();
        }
        unsigned int onesBefore = sscan[t] - bit;
        unsigned int totOnes = sscan[255];
        unsigned int pos = bit ? (256u - totOnes + onesBefore)
                               : ((unsigned)t - onesBefore);
        skey[pos] = k; spay[pos] = p; __syncthreads();
        k = skey[t]; p = spay[t]; __syncthreads();
    }
    unsigned int digit = (k >> shift) & 255u;
    if (t == 0) sscan[digit] = 0u;
    else {
        unsigned int pd = (skey[t - 1] >> shift) & 255u;
        if (pd != digit) sscan[digit] = (unsigned)t;      // digit start in block
    }
    __syncthreads();
    if (k != 0xFFFFFFFFu) {
        unsigned int local = (unsigned)t - sscan[digit];
        unsigned int pos = hist[(size_t)blockIdx.x * 256 + digit] + local;
        keyOut[pos] = k; payOut[pos] = p;
    }
}

// ---------------- normalize + fp16 convert ----------------

__global__ void k_xnorm(const float* x, _Float16* xn) {
    __shared__ float s[128];
    int t = threadIdx.x, row = blockIdx.x;
    float v = x[(size_t)row * DIM + t];
    s[t] = v * v; __syncthreads();
    for (int off = 64; off > 0; off >>= 1) {
        if (t < off) s[t] += s[t + off];
        __syncthreads();
    }
    float sc = 1.0f / fmaxf(sqrtf(s[0]), 1e-12f);
    xn[(size_t)row * DIM + t] = (_Float16)(v * sc);
}

__global__ void k_wnorm(const float* weights, const int* positive,
                        const unsigned int* negs, const int* d_n, _Float16* wn) {
    __shared__ float s[128];
    int t = threadIdx.x, col = blockIdx.x;
    if (col >= WTOT) {                               // zero the pad columns
        wn[(size_t)col * DIM + t] = (_Float16)0.0f;
        return;                                      // block-uniform exit
    }
    int n = *d_n;
    int cls = (col < n) ? positive[col] : (int)negs[col - n];
    float v = weights[(size_t)cls * DIM + t];
    s[t] = v * v; __syncthreads();
    for (int off = 64; off > 0; off >>= 1) {
        if (t < off) s[t] += s[t + off];
        __syncthreads();
    }
    float sc = 1.0f / fmaxf(sqrtf(s[0]), 1e-12f);
    wn[(size_t)col * DIM + t] = (_Float16)(v * sc);
}

// ---------------- WMMA GEMM: out[4096][100000] = Xn @ Wn^T ----------------
// Block = 256 threads = 8 wave32. TDM stages the shared 64x128-half W tile
// (16 KB) into LDS once per block; A fragments come straight from global (L2-
// resident, disjoint per wave). Compute loop is branch-free (Wn padded).

__global__ void __launch_bounds__(256) k_gemm(const _Float16* __restrict__ xn,
                                              const _Float16* __restrict__ wn,
                                              float* __restrict__ out) {
    __shared__ _Float16 lb[64 * DIM];                // W tile: 64 cols x 128 halfs

    int lane = threadIdx.x & 31;
    int wave = threadIdx.x >> 5;
    int mBase = blockIdx.y * 128 + wave * 16;
    int nBase = blockIdx.x * 64;

    // ---- TDM: async DMA of W tile into LDS (wave 0 issues) ----
    if (wave == 0) {
        unsigned long long ga =
            (unsigned long long)(uintptr_t)(wn + (size_t)nBase * DIM);
        unsigned int lds = (unsigned int)(uintptr_t)&lb[0];
        u32x4 g0;
        g0[0] = 1u;                                   // count=1, user mode
        g0[1] = lds;                                  // lds_addr
        g0[2] = (unsigned int)(ga & 0xFFFFFFFFu);     // global_addr[31:0]
        g0[3] = (unsigned int)((ga >> 32) & 0x01FFFFFFu) | (2u << 30); // [56:32]|type=2
        i32x8 g1;
        g1[0] = (1 << 16);                            // data_size=1 (2B), mask=0
        g1[1] = (DIM & 0xFFFF) << 16;                 // tensor_dim0[15:0]
        g1[2] = (DIM >> 16) | ((WPAD & 0xFFFF) << 16);// dim0[31:16] | dim1[15:0]
        g1[3] = (WPAD >> 16) | (DIM << 16);           // dim1[31:16] | tile_dim0=128
        g1[4] = 64;                                   // tile_dim1=64, tile_dim2=0
        g1[5] = DIM;                                  // tensor_dim0_stride[31:0]
        g1[6] = 0;
        g1[7] = 0;
        i32x4 z4 = {0, 0, 0, 0};
        i32x8 z8 = {0, 0, 0, 0, 0, 0, 0, 0};
        __builtin_amdgcn_tensor_load_to_lds(g0, g1, z4, z4, z8, 0);
    }

    // ---- A fragments from global, overlapped with the TDM transfer ----
    // A k order chosen here (B matches): lanes0-15 K=0..15, lanes16-31 K=16..31
    int kh = lane >> 4;
    const _Float16* aptr = xn + (size_t)(mBase + (lane & 15)) * DIM + kh * 16;
    __builtin_prefetch(aptr, 0, 0);                   // global_prefetch_b8
    v16h a[4];
#pragma unroll
    for (int kk = 0; kk < 4; ++kk) {
        v8h alo = *(const v8h*)(aptr + kk * 32);
        v8h ahi = *(const v8h*)(aptr + kk * 32 + 8);
        a[kk] = __builtin_shufflevector(alo, ahi, 0, 1, 2, 3, 4, 5, 6, 7,
                                                  8, 9, 10, 11, 12, 13, 14, 15);
    }

    if (wave == 0) __builtin_amdgcn_s_wait_tensorcnt(0);
    __syncthreads();                                  // W tile visible to all waves

    // ---- branch-free WMMA loop, B fragments from LDS ----
    // B k order chosen here (matches A): lanes0-15 K=0..15, lanes16-31 K=16..31; N=lane&15
    int ncl = lane & 15;
    int kb  = kh * 16;
    v8f acc0 = {}, acc1 = {}, acc2 = {}, acc3 = {};
#pragma unroll
    for (int kk = 0; kk < 4; ++kk) {
        int k0 = kk * 32;
        v16h b0 = *(const v16h*)(lb + (size_t)( 0 + ncl) * DIM + kb + k0);
        v16h b1 = *(const v16h*)(lb + (size_t)(16 + ncl) * DIM + kb + k0);
        v16h b2 = *(const v16h*)(lb + (size_t)(32 + ncl) * DIM + kb + k0);
        v16h b3 = *(const v16h*)(lb + (size_t)(48 + ncl) * DIM + kb + k0);
        acc0 = __builtin_amdgcn_wmma_f32_16x16x32_f16(false, a[kk], false, b0,
                                                      (short)0, acc0, false, false);
        acc1 = __builtin_amdgcn_wmma_f32_16x16x32_f16(false, a[kk], false, b1,
                                                      (short)0, acc1, false, false);
        acc2 = __builtin_amdgcn_wmma_f32_16x16x32_f16(false, a[kk], false, b2,
                                                      (short)0, acc2, false, false);
        acc3 = __builtin_amdgcn_wmma_f32_16x16x32_f16(false, a[kk], false, b3,
                                                      (short)0, acc3, false, false);
    }

    // ---- guarded stores (only last x-block has pad columns) ----
    // C/D layout: VGPR r -> row mBase + (lane>>4)*8 + r, col = sub + (lane&15)
    bool on0 = (nBase +  0) < WTOT;
    bool on1 = (nBase + 16) < WTOT;
    bool on2 = (nBase + 32) < WTOT;
    bool on3 = (nBase + 48) < WTOT;
    int rrow = mBase + kh * 8;
#pragma unroll
    for (int r = 0; r < 8; ++r) {
        size_t ro = (size_t)(rrow + r) * WTOT;
        if (on0) out[ro + nBase +  0 + ncl] = acc0[r];
        if (on1) out[ro + nBase + 16 + ncl] = acc1[r];
        if (on2) out[ro + nBase + 32 + ncl] = acc2[r];
        if (on3) out[ro + nBase + 48 + ncl] = acc3[r];
    }
}

// ---------------- launch ----------------

extern "C" void kernel_launch(void* const* d_in, const int* in_sizes, int n_in,
                              void* d_out, int out_size, void* d_ws, size_t ws_size,
                              hipStream_t stream) {
    (void)in_sizes; (void)n_in; (void)out_size; (void)ws_size;
    const float* x       = (const float*)d_in[0];
    const float* weights = (const float*)d_in[1];
    const float* perm    = (const float*)d_in[2];
    const int*   y       = (const int*)d_in[3];
    float* out = (float*)d_out;

    auto al = [](size_t v) { return (v + 255) & ~(size_t)255; };
    uint8_t* p = (uint8_t*)d_ws;
    unsigned int* flags = (unsigned int*)p; p += al((size_t)NC * 4);
    int*          idx_map = (int*)p;        p += al((size_t)NC * 4);
    unsigned int* keyA = (unsigned int*)p;  p += al((size_t)NC * 4);
    unsigned int* keyB = (unsigned int*)p;  p += al((size_t)NC * 4);
    unsigned int* payA = (unsigned int*)p;  p += al((size_t)NC * 4);
    unsigned int* payB = (unsigned int*)p;  p += al((size_t)NC * 4);
    unsigned int* bsum = (unsigned int*)p;  p += al((size_t)NB * 4);
    int*          d_n  = (int*)p;           p += 256;
    unsigned int* hist = (unsigned int*)p;  p += al((size_t)NB * 256 * 4);
    int*          positive = (int*)p;       p += al((size_t)BATCH * 4);
    _Float16*     xn = (_Float16*)p;        p += al((size_t)BATCH * DIM * 2);
    _Float16*     wn = (_Float16*)p;        p += al((size_t)WPAD * DIM * 2);

    // unique(y) -> positive[], idx_map[], n, remap_y
    k_clear<<<NB, 256, 0, stream>>>(flags);
    k_mark<<<(BATCH + 255) / 256, 256, 0, stream>>>(y, flags);
    k_block_reduce<<<NB, 256, 0, stream>>>(flags, bsum);
    k_scan_bsums<<<1, 256, 0, stream>>>(bsum, d_n);
    k_emit_pos<<<NB, 256, 0, stream>>>(flags, bsum, positive, idx_map);
    k_remap<<<(BATCH + 255) / 256, 256, 0, stream>>>(y, idx_map,
                                                     out + (size_t)BATCH * WTOT);

    // descending stable sort of masked perm -> negatives in top_k order
    k_init_keys<<<NB, 256, 0, stream>>>(perm, flags, keyA, payA);
    unsigned int *ki = keyA, *pi = payA, *ko = keyB, *po = payB;
    for (int pass = 0; pass < 4; ++pass) {
        int shift = pass * 8;
        k_hist<<<NB, 256, 0, stream>>>(ki, hist, shift);
        k_scan_hist<<<1, 256, 0, stream>>>(hist);
        k_scatter<<<NB, 256, 0, stream>>>(ki, pi, ko, po, hist, shift);
        unsigned int* t;
        t = ki; ki = ko; ko = t;
        t = pi; pi = po; po = t;
    }
    // 4 passes -> sorted result back in keyA/payA (ki/pi == keyA/payA)

    // normalize to fp16 operands (wn padded to WPAD columns)
    k_xnorm<<<BATCH, 128, 0, stream>>>(x, xn);
    k_wnorm<<<WPAD, 128, 0, stream>>>(weights, positive, pi, d_n, wn);

    // logits GEMM via WMMA + TDM-staged W tiles
    dim3 grid(WPAD / 64, BATCH / 128);
    k_gemm<<<grid, 256, 0, stream>>>(xn, wn, out);
}
